// Attention_19344532701733
// MI455X (gfx1250) — matrix operand order repaired and checked by reading, not compiled, not run on hardware
//
#include <hip/hip_runtime.h>
#include <hip/hip_bf16.h>
#include <math.h>

// ---------------------------------------------------------------------------
// CDNA5 (gfx1250) causal multi-head attention, bf16 WMMA everywhere.
// B=2, N=2048, D=1024, H=16, Dh=64.
// Round 2: flash-attention K/V tiles staged to LDS via
// global_load_async_to_lds_b128 (ASYNCcnt path), shared by all 4 waves.
// ---------------------------------------------------------------------------

typedef __bf16 bf16;
typedef __attribute__((ext_vector_type(16))) __bf16 v16bf;
typedef __attribute__((ext_vector_type(8)))  __bf16 v8bf;
typedef __attribute__((ext_vector_type(8)))  float  v8f;

#define BATCH   2
#define NSEQ    2048
#define DMODEL  1024
#define HEADS   16
#define DHEAD   64
#define NINNER  (HEADS * DHEAD)          // 1024
#define MASKVAL (-1.7014118e38f)         // -(FLT_MAX/2)

// ---- WMMA wrapper: D = A(16x32 bf16) * B(32x16 bf16) + C(16x16 f32) -------
__device__ __forceinline__ v8f wmma_bf16(v16bf a, v16bf b, v8f c) {
    return __builtin_amdgcn_wmma_f32_16x16x32_bf16(
        /*neg_a=*/false, a, /*neg_b=*/false, b,
        /*c_mod=*/(short)0, c, /*reuse_a=*/false, /*reuse_b=*/false);
}

// ---- A-fragment (16x32, row-major source, leading dim ld in elements) -----
// ISA layout: lanes 0-15 -> row m=lane, K {0..7, 16..23}; lanes 16-31 -> row
// m=lane-16, K {8..15, 24..31}.  Two 16-byte loads per lane.
__device__ __forceinline__ v16bf load_frag_a(const bf16* p, int ld, int lane) {
    const bf16* base = p + (size_t)(lane & 15) * ld + ((lane >> 4) << 3);
    v8bf lo = *(const v8bf*)(base);
    v8bf hi = *(const v8bf*)(base + 16);
    v16bf r;
#pragma unroll
    for (int i = 0; i < 8; ++i) { r[i] = lo[i]; r[i + 8] = hi[i]; }
    return r;
}

// ---- B-fragment (32x16) from a TRANSPOSED source: pT is [N x K] row-major,
// i.e. column-major B.  ISA layout: lane holds column n=lane%16, K
// {0..7, 16..23} for lanes 0-15 and {8..15, 24..31} for lanes 16-31.
__device__ __forceinline__ v16bf load_frag_b(const bf16* pT, int ld, int lane) {
    const bf16* base = pT + (size_t)(lane & 15) * ld + ((lane >> 4) << 3);
    v8bf lo = *(const v8bf*)(base);
    v8bf hi = *(const v8bf*)(base + 16);
    v16bf r;
#pragma unroll
    for (int i = 0; i < 8; ++i) { r[i] = lo[i]; r[i + 8] = hi[i]; }
    return r;
}

// ---- gfx1250 async copy: global -> LDS, 16 bytes per lane -----------------
__device__ __forceinline__ void async_g2l_b128(unsigned lds_off, const bf16* g) {
    asm volatile("global_load_async_to_lds_b128 %0, %1, off"
                 :: "v"(lds_off), "v"(g) : "memory");
}
__device__ __forceinline__ void wait_async0() {
    asm volatile("s_wait_asynccnt 0" ::: "memory");
}

// ---------------------------------------------------------------------------
// Stage 0: fp32 -> bf16 converts (straight + transposing)
// ---------------------------------------------------------------------------
__global__ void cvt_bf16(const float* __restrict__ in, bf16* __restrict__ out, int n) {
    int i = blockIdx.x * blockDim.x + threadIdx.x;
    if (i < n) out[i] = (bf16)in[i];
}

// w: [K x N] row-major  ->  wT: [N x K] row-major (bf16)
__global__ void cvt_transpose_bf16(const float* __restrict__ w, bf16* __restrict__ wT,
                                   int K, int N) {
    int i = blockIdx.x * blockDim.x + threadIdx.x;
    if (i < K * N) {
        int k = i / N, n = i - k * N;
        wT[(size_t)n * K + k] = (bf16)w[i];
    }
}

// ---------------------------------------------------------------------------
// Stage 1: QKV projection.  C[4096,3072] = Xbf[4096,1024] @ Wqkv.
// 8 waves/WG (2x4), wave tile 32x64, WG tile 64x256.
// Q scaled by Dh^-0.5 at writeout; K row-major [B,H,N,Dh]; V transposed
// [B,H,Dh,N] so the attention PV GEMM gets contiguous B-fragments.
// ---------------------------------------------------------------------------
__global__ __launch_bounds__(256) void qkv_gemm(const bf16* __restrict__ X,
                                                const bf16* __restrict__ WT,
                                                bf16* __restrict__ Qb,
                                                bf16* __restrict__ Kb,
                                                bf16* __restrict__ Vt) {
    const int lane = threadIdx.x & 31;
    const int wave = threadIdx.x >> 5;
    const int m0 = blockIdx.x * 64 + (wave >> 2) * 32;
    const int n0 = blockIdx.y * 256 + (wave & 3) * 64;

    v8f acc[2][4] = {};
    for (int k0 = 0; k0 < DMODEL; k0 += 32) {
        v16bf a0 = load_frag_a(X + (size_t)m0 * DMODEL + k0, DMODEL, lane);
        v16bf a1 = load_frag_a(X + (size_t)(m0 + 16) * DMODEL + k0, DMODEL, lane);
#pragma unroll
        for (int j = 0; j < 4; ++j) {
            v16bf b = load_frag_b(WT + (size_t)(n0 + j * 16) * DMODEL + k0, DMODEL, lane);
            acc[0][j] = wmma_bf16(a0, b, acc[0][j]);
            acc[1][j] = wmma_bf16(a1, b, acc[1][j]);
        }
    }
#pragma unroll
    for (int mi = 0; mi < 2; ++mi)
#pragma unroll
        for (int j = 0; j < 4; ++j)
#pragma unroll
            for (int i = 0; i < 8; ++i) {
                int m = m0 + mi * 16 + i + ((lane >> 4) << 3);   // row of [4096]
                int n = n0 + j * 16 + (lane & 15);               // col of [3072]
                float v = acc[mi][j][i];
                int which = n >> 10, rem = n & 1023;
                int h = rem >> 6, d = rem & 63;
                int bb = m >> 11, t = m & (NSEQ - 1);
                size_t bh = (size_t)(bb * HEADS + h);
                if (which == 0)
                    Qb[(bh * NSEQ + t) * DHEAD + d] = (bf16)(v * 0.125f); // Dh^-0.5
                else if (which == 1)
                    Kb[(bh * NSEQ + t) * DHEAD + d] = (bf16)v;
                else
                    Vt[(bh * DHEAD + d) * NSEQ + t] = (bf16)v;            // transposed
            }
}

// ---------------------------------------------------------------------------
// Stage 2: flash attention.  blockIdx.x = b*H+h (32), blockIdx.y = q-block of
// 64 (32).  4 waves/WG; each wave owns 16 queries, streams 64-key tiles with
// online softmax.  K and V^T tiles are staged once per WG into LDS with
// global_load_async_to_lds_b128 (ASYNCcnt) and shared by all waves; S = Q K^T
// and O += P V both via WMMA; P is rotated from C-layout to A-layout through
// a per-wave LDS tile (wave lockstep + dscnt drain, no barrier).
// The kv loop bound is uniform across the WG ((blockIdx.y+1)*64); waves run
// at most one fully-masked extra tile, which contributes exactly zero.
// ---------------------------------------------------------------------------
__global__ __launch_bounds__(128) void flash_attn(const bf16* __restrict__ Qb,
                                                  const bf16* __restrict__ Kb,
                                                  const bf16* __restrict__ Vt,
                                                  bf16* __restrict__ Ob) {
    __shared__ __align__(16) bf16 Kt[64 * 64];        // 8 KB  [key][dim]
    __shared__ __align__(16) bf16 Vts[64 * 64];       // 8 KB  [dim][key]
    __shared__ __align__(16) bf16 Pt[4][16 * 64];     // 8 KB  per-wave P tiles

    const int lane = threadIdx.x & 31;
    const int wave = threadIdx.x >> 5;
    const int bh = blockIdx.x;
    const int q0 = blockIdx.y * 64 + wave * 16;

    const bf16* Qh = Qb + (size_t)bh * NSEQ * DHEAD;
    const bf16* Kh = Kb + (size_t)bh * NSEQ * DHEAD;
    const bf16* Vh = Vt + (size_t)bh * DHEAD * NSEQ;

    // Q fragments for this wave's 16 rows (Dh=64 -> 2 A-frags), loaded once.
    v16bf qa0 = load_frag_a(Qh + (size_t)q0 * DHEAD + 0, DHEAD, lane);
    v16bf qa1 = load_frag_a(Qh + (size_t)q0 * DHEAD + 32, DHEAD, lane);

    v8f o[4] = {};
    float mrow[8], lrow[8];
#pragma unroll
    for (int i = 0; i < 8; ++i) { mrow[i] = -3.0e38f; lrow[i] = 0.0f; }

    bf16* pw = &Pt[wave][0];
    const unsigned kt_off = (unsigned)(size_t)(void*)&Kt[0];
    const unsigned vt_off = (unsigned)(size_t)(void*)&Vts[0];

    const int kv_lim = (blockIdx.y + 1) * 64;         // uniform across the WG

    for (int kv0 = 0; kv0 < kv_lim; kv0 += 64) {
        // ---- async-stage K (contiguous 8 KB) and V^T (64 rows x 128 B) ---
        // 512 chunks of 16 B each per tile; 128 threads x 4 chunks.
#pragma unroll
        for (int it = 0; it < 4; ++it) {
            int c = threadIdx.x + it * 128;           // 0..511
            async_g2l_b128(kt_off + c * 16, Kh + (size_t)kv0 * DHEAD + c * 8);
            int r = c >> 3, c8 = (c & 7) * 8;
            async_g2l_b128(vt_off + c * 16, Vh + (size_t)r * NSEQ + kv0 + c8);
        }
        wait_async0();
        __syncthreads();

        // ---- S tile 16x64 = Q (16x64) . K^T, B-frags from LDS ------------
        v8f s[4] = {};
#pragma unroll
        for (int j = 0; j < 4; ++j) {
            const bf16* kp = Kt + (size_t)(j * 16) * DHEAD;
            v16bf b0 = load_frag_b(kp + 0, DHEAD, lane);
            v16bf b1 = load_frag_b(kp + 32, DHEAD, lane);
            s[j] = wmma_bf16(qa0, b0, s[j]);
            s[j] = wmma_bf16(qa1, b1, s[j]);
        }
        // ---- causal mask (boundary / fully-masked tiles) -----------------
        if (kv0 + 63 > q0) {
#pragma unroll
            for (int j = 0; j < 4; ++j) {
                int n = kv0 + j * 16 + (lane & 15);
#pragma unroll
                for (int i = 0; i < 8; ++i) {
                    int m = q0 + i + ((lane >> 4) << 3);
                    if (n > m) s[j][i] = MASKVAL;
                }
            }
        }
        // ---- online softmax update ---------------------------------------
        float tmax[8];
#pragma unroll
        for (int i = 0; i < 8; ++i) {
            float t = s[0][i];
            t = fmaxf(t, s[1][i]); t = fmaxf(t, s[2][i]); t = fmaxf(t, s[3][i]);
#pragma unroll
            for (int off = 1; off < 16; off <<= 1)
                t = fmaxf(t, __shfl_xor(t, off, 32));
            tmax[i] = t;
        }
        float mnew[8], alpha[8], tsum[8];
#pragma unroll
        for (int i = 0; i < 8; ++i) {
            mnew[i] = fmaxf(mrow[i], tmax[i]);
            alpha[i] = __expf(mrow[i] - mnew[i]);
            tsum[i] = 0.0f;
        }
#pragma unroll
        for (int j = 0; j < 4; ++j)
#pragma unroll
            for (int i = 0; i < 8; ++i) {
                float p = __expf(s[j][i] - mnew[i]);
                tsum[i] += p;
                // C-layout element (m=i+(lane/16)*8, n=j*16+lane%16) -> LDS row-major
                pw[(i + ((lane >> 4) << 3)) * 64 + j * 16 + (lane & 15)] = (bf16)p;
            }
#pragma unroll
        for (int i = 0; i < 8; ++i) {
#pragma unroll
            for (int off = 1; off < 16; off <<= 1)
                tsum[i] += __shfl_xor(tsum[i], off, 32);
            lrow[i] = lrow[i] * alpha[i] + tsum[i];
            mrow[i] = mnew[i];
        }
#pragma unroll
        for (int j = 0; j < 4; ++j)
#pragma unroll
            for (int i = 0; i < 8; ++i) o[j][i] *= alpha[i];

        // wave-private LDS P tile: lockstep wave needs only a dscnt drain.
        asm volatile("s_wait_dscnt 0" ::: "memory");

        v16bf pa0 = load_frag_a(pw + 0, 64, lane);
        v16bf pa1 = load_frag_a(pw + 32, 64, lane);
        // ---- O += P (16x64) . V (64x64); V^T tile in LDS -----------------
#pragma unroll
        for (int j = 0; j < 4; ++j) {
            const bf16* vp = Vts + (size_t)(j * 16) * 64;
            v16bf vb0 = load_frag_b(vp + 0, 64, lane);
            v16bf vb1 = load_frag_b(vp + 32, 64, lane);
            o[j] = wmma_bf16(pa0, vb0, o[j]);
            o[j] = wmma_bf16(pa1, vb1, o[j]);
        }
        __syncthreads();   // all waves done reading K/V tiles before restage
    }

    // ---- finalize + write O as [B, N, H*Dh] bf16 -------------------------
    int bb = bh >> 4, h = bh & 15;
#pragma unroll
    for (int j = 0; j < 4; ++j)
#pragma unroll
        for (int i = 0; i < 8; ++i) {
            int m = q0 + i + ((lane >> 4) << 3);
            int n = h * DHEAD + j * 16 + (lane & 15);
            Ob[((size_t)(bb * NSEQ + m)) * NINNER + n] = (bf16)(o[j][i] / lrow[i]);
        }
}

// ---------------------------------------------------------------------------
// Stage 3: out projection.  out[4096,1024] = O[4096,1024] @ Wout + b_out (f32)
// ---------------------------------------------------------------------------
__global__ __launch_bounds__(256) void out_gemm(const bf16* __restrict__ O,
                                                const bf16* __restrict__ WT,
                                                const float* __restrict__ bias,
                                                float* __restrict__ out) {
    const int lane = threadIdx.x & 31;
    const int wave = threadIdx.x >> 5;
    const int m0 = blockIdx.x * 64 + (wave >> 2) * 32;
    const int n0 = blockIdx.y * 256 + (wave & 3) * 64;

    v8f acc[2][4] = {};
    for (int k0 = 0; k0 < NINNER; k0 += 32) {
        v16bf a0 = load_frag_a(O + (size_t)m0 * NINNER + k0, NINNER, lane);
        v16bf a1 = load_frag_a(O + (size_t)(m0 + 16) * NINNER + k0, NINNER, lane);
#pragma unroll
        for (int j = 0; j < 4; ++j) {
            v16bf b = load_frag_b(WT + (size_t)(n0 + j * 16) * NINNER + k0, NINNER, lane);
            acc[0][j] = wmma_bf16(a0, b, acc[0][j]);
            acc[1][j] = wmma_bf16(a1, b, acc[1][j]);
        }
    }
#pragma unroll
    for (int mi = 0; mi < 2; ++mi)
#pragma unroll
        for (int j = 0; j < 4; ++j) {
            int n = n0 + j * 16 + (lane & 15);
            float bv = bias[n];
#pragma unroll
            for (int i = 0; i < 8; ++i) {
                int m = m0 + mi * 16 + i + ((lane >> 4) << 3);
                out[(size_t)m * DMODEL + n] = acc[mi][j][i] + bv;
            }
        }
}

// ---------------------------------------------------------------------------
extern "C" void kernel_launch(void* const* d_in, const int* in_sizes, int n_in,
                              void* d_out, int out_size, void* d_ws, size_t ws_size,
                              hipStream_t stream) {
    const float* x     = (const float*)d_in[0];   // [2,2048,1024]
    const float* w_qkv = (const float*)d_in[1];   // [1024,3072]
    const float* w_out = (const float*)d_in[2];   // [1024,1024]
    const float* b_out = (const float*)d_in[3];   // [1024]
    float* out = (float*)d_out;

    char* ws = (char*)d_ws;                       // 48 MiB used
    bf16* Xb  = (bf16*)(ws);                      //  8 MiB  x in bf16
    bf16* WqT = (bf16*)(ws + (8u  << 20));        //  6 MiB  w_qkv^T bf16
    bf16* WoT = (bf16*)(ws + (14u << 20));        //  2 MiB  w_out^T bf16
    bf16* Qb  = (bf16*)(ws + (16u << 20));        //  8 MiB  [B,H,N,Dh]
    bf16* Kb  = (bf16*)(ws + (24u << 20));        //  8 MiB  [B,H,N,Dh]
    bf16* Vt  = (bf16*)(ws + (32u << 20));        //  8 MiB  [B,H,Dh,N]
    bf16* Ob  = (bf16*)(ws + (40u << 20));        //  8 MiB  [B,N,H*Dh]

    const int nx = BATCH * NSEQ * DMODEL;         // 4,194,304
    cvt_bf16<<<(nx + 255) / 256, 256, 0, stream>>>(x, Xb, nx);
    cvt_transpose_bf16<<<(DMODEL * 3 * NINNER + 255) / 256, 256, 0, stream>>>(
        w_qkv, WqT, DMODEL, 3 * NINNER);
    cvt_transpose_bf16<<<(NINNER * DMODEL + 255) / 256, 256, 0, stream>>>(
        w_out, WoT, NINNER, DMODEL);

    qkv_gemm<<<dim3((BATCH * NSEQ) / 64, (3 * NINNER) / 256), 256, 0, stream>>>(
        Xb, WqT, Qb, Kb, Vt);

    flash_attn<<<dim3(BATCH * HEADS, NSEQ / 64), 128, 0, stream>>>(Qb, Kb, Vt, Ob);

    out_gemm<<<dim3((BATCH * NSEQ) / 64, DMODEL / 256), 256, 0, stream>>>(
        Ob, WoT, b_out, out);
}
